// BreakthroughSNN_10496900071679
// MI455X (gfx1250) — matrix operand order repaired and checked by reading, compiled, not run on hardware
//
#include <hip/hip_runtime.h>
#include <hip/hip_bf16.h>
#include <math.h>

// ---------------------------------------------------------------------------
// Spiking-SSM for MI455X (gfx1250, wave32, WMMA + TDM).
// All matmul activations are binary spikes (exact in f16); weights cast to
// f16 once; all GEMMs run on v_wmma_f32_16x16x32_f16 with f32 accumulate.
// The vocab projection stages Wp tiles into LDS via the Tensor Data Mover
// (double buffered, TDM pad feature used for bank-conflict-free LDS stride).
// Workspace use ~126 MB (output itself is 262 MB).
// ---------------------------------------------------------------------------

typedef __attribute__((ext_vector_type(16))) _Float16 v16h;
typedef __attribute__((ext_vector_type(8)))  _Float16 v8h;
typedef __attribute__((ext_vector_type(8)))  float    v8f;
typedef __attribute__((ext_vector_type(4)))  unsigned v4u;
typedef __attribute__((ext_vector_type(4)))  int      v4i;
typedef __attribute__((ext_vector_type(8)))  int      v8i;
typedef __attribute__((ext_vector_type(4)))  unsigned uint4v;

#define TS     20
#define NLAYER 4
#define DM     512
#define DS     128
#define VOC    32000
#define MB     4
#define SEQ    512
#define MROWS  (MB * SEQ)        /* 2048 flattened (b,s) rows */
#define MT     64                /* rows per workgroup in step kernel */
#define SH_LD  136               /* padded f16 stride for LDS h tile   */

#if __has_builtin(__builtin_amdgcn_tensor_load_to_lds)
#define USE_TDM 1
#else
#define USE_TDM 0
#endif

// ---- WMMA fragment loaders (per CDNA5 ISA VGPR layouts) --------------------
// A (16x32 f16, MxK): lane L (L<16) holds row M=L, K in {0..7} then {16..23};
// lane L+16 holds same row, K in {8..15} then {24..31}.  With A row-major
// this is two contiguous 16B loads per lane.
__device__ __forceinline__ v16h frag_a16(const _Float16* p, int ld) {
  const int lane = threadIdx.x & 31;
  const _Float16* q = p + (lane & 15) * ld + ((lane >> 4) << 3);
  v8h lo = *reinterpret_cast<const v8h*>(q);
  v8h hi = *reinterpret_cast<const v8h*>(q + 16);
  return __builtin_shufflevector(lo, hi, 0, 1, 2, 3, 4, 5, 6, 7,
                                         8, 9, 10, 11, 12, 13, 14, 15);
}
// B (32x16 f16, KxN): lane n (n<16) holds column N=n, K in {0..7} then
// {16..23}; lane n+16 holds K in {8..15} then {24..31}.  With B stored
// transposed row-major (N x K) this is two contiguous 16B loads per lane.
__device__ __forceinline__ v16h frag_b16(const _Float16* p, int ld) {
  const int lane = threadIdx.x & 31;
  const _Float16* q = p + (lane & 15) * ld + ((lane >> 4) << 3);
  v8h lo = *reinterpret_cast<const v8h*>(q);
  v8h hi = *reinterpret_cast<const v8h*>(q + 16);
  return __builtin_shufflevector(lo, hi, 0, 1, 2, 3, 4, 5, 6, 7,
                                         8, 9, 10, 11, 12, 13, 14, 15);
}
__device__ __forceinline__ v8f wmma16(v16h a, v16h b, v8f c) {
  return __builtin_amdgcn_wmma_f32_16x16x32_f16(false, a, false, b,
                                                (short)0, c, false, false);
}

// ---- TDM 2D tile load: global (row-major, stride elems) -> LDS -------------
// D# per CDNA5 ISA ch.8: group0 = {count=1, lds_addr, global_addr, type=2};
// group1 = {flags, tensor dims, tile dims, dim0 stride}. 2D only: groups 2/3
// zero (tile_dim2/3 = 0 -> unused).  Toolchain builtin is the 6-arg form:
// (v4u g0, v8i g1, v4i g2, v4i g3, v8i g4, i32 cpol).
__device__ __forceinline__ void tdm_load_2d(unsigned lds_addr,
                                            const void* gaddr,
                                            unsigned tile_d0, unsigned tile_d1,
                                            unsigned tens_d0, unsigned tens_d1,
                                            unsigned stride_el,
                                            unsigned flags_w0) {
#if USE_TDM
  unsigned long long ga = (unsigned long long)(uintptr_t)gaddr;
  v4u g0 = {1u,                                   // count=1, user descriptor
            lds_addr,                             // LDS byte address
            (unsigned)(ga & 0xffffffffu),         // global_addr[31:0]
            (unsigned)((ga >> 32) & 0x01ffffffu) | 0x80000000u}; // [56:32]|type=2
  v8i g1 = {(int)flags_w0,                                    // mask/ds/pad
            (int)((tens_d0 & 0xffffu) << 16),                 // dim0[15:0]
            (int)((tens_d0 >> 16) | ((tens_d1 & 0xffffu) << 16)),
            (int)((tens_d1 >> 16) | (tile_d0 << 16)),
            (int)(tile_d1 & 0xffffu),                         // tile_dim1,2=0
            (int)stride_el, 0, 0};                            // dim0_stride
  v4i gz4 = {0, 0, 0, 0};
  v8i gz8 = {0, 0, 0, 0, 0, 0, 0, 0};
  __builtin_amdgcn_tensor_load_to_lds(g0, g1, gz4, gz4, gz8, 0);
#endif
}

// ---- small utility kernels -------------------------------------------------
__global__ void cvt_f32_f16_kernel(const float* __restrict__ src,
                                   _Float16* __restrict__ dst, int n) {
  int i = blockIdx.x * blockDim.x + threadIdx.x;
  if (i < n) dst[i] = (_Float16)src[i];
}

// embedding gather + temporal one-hot encode -> spike tensor (TS, M, DM) f16
__global__ void encode_kernel(const int* __restrict__ ids,
                              const float* __restrict__ emb,
                              _Float16* __restrict__ X) {
  int i = blockIdx.x * blockDim.x + threadIdx.x;
  if (i >= MROWS * DM) return;
  int m = i / DM, d = i - m * DM;
  float e = emb[(size_t)ids[m] * DM + d];
  float sg = 1.0f / (1.0f + __expf(-e));
  int idx = (int)floorf(sg * (float)(TS - 1));
  idx = idx < 0 ? 0 : (idx > TS - 1 ? TS - 1 : idx);
#pragma unroll
  for (int t = 0; t < TS; ++t)
    X[(size_t)t * (MROWS * DM) + i] = (_Float16)((t == idx) ? 1.0f : 0.0f);
}

// sum spikes over T -> integer counts (exact in f16)
__global__ void reduce_kernel(const _Float16* __restrict__ X,
                              _Float16* __restrict__ cnt) {
  int i = blockIdx.x * blockDim.x + threadIdx.x;
  if (i >= MROWS * DM) return;
  float s = 0.0f;
#pragma unroll
  for (int t = 0; t < TS; ++t) s += (float)X[(size_t)t * (MROWS * DM) + i];
  cnt[i] = (_Float16)s;
}

// ---- fused SSM step: GEMM1 -> LIF1 -> GEMM2 -> LIF2 ------------------------
// grid.x = MROWS/MT = 32 workgroups, 256 threads = 8 waves.
__global__ __launch_bounds__(256)
void snn_step_kernel(const _Float16* __restrict__ xin,   // (M, DM) spikes at t
                     _Float16* __restrict__ yout,        // (M, DM) out spikes
                     const _Float16* __restrict__ Aw,    // (DS, DS)  = B-op
                     const _Float16* __restrict__ Bw,    // (DS, DM)  = B-op
                     const _Float16* __restrict__ Cw,    // (DM, DS)  = B-op
                     const float* __restrict__ Dv,       // (DM,)
                     float* __restrict__ v1,             // (M, DS)
                     float* __restrict__ v2,             // (M, DM)
                     _Float16* __restrict__ hbuf,        // (M, DS) spikes
                     int first) {
  __shared__ _Float16 sh_h[MT * SH_LD];

  const int lane = threadIdx.x & 31;
  const int wv   = threadIdx.x >> 5;
  const int mi   = wv & 3;
  const int nh   = wv >> 2;                 // 0..1
  const int m0   = blockIdx.x * MT;
  const int mrow0 = m0 + mi * 16;
  const int mloc_base = mi * 16 + ((lane >> 4) << 3);
  const int mg_base   = m0 + mloc_base;
  const v8f vzero = {0.f, 0.f, 0.f, 0.f, 0.f, 0.f, 0.f, 0.f};

  // ---------------- GEMM1: SU = h_prev @ A^T + x_t @ B^T  (MT x 128) -------
  v8f acc[4];
#pragma unroll
  for (int j = 0; j < 4; ++j) acc[j] = vzero;

  if (!first) {
#pragma unroll
    for (int kk = 0; kk < DS; kk += 32) {
      v16h a = frag_a16(hbuf + (size_t)mrow0 * DS + kk, DS);
#pragma unroll
      for (int j = 0; j < 4; ++j) {
        v16h b = frag_b16(Aw + (size_t)((nh * 4 + j) * 16) * DS + kk, DS);
        acc[j] = wmma16(a, b, acc[j]);
      }
    }
  }
#pragma unroll 4
  for (int kk = 0; kk < DM; kk += 32) {
    v16h a = frag_a16(xin + (size_t)mrow0 * DM + kk, DM);
#pragma unroll
    for (int j = 0; j < 4; ++j) {
      v16h b = frag_b16(Bw + (size_t)((nh * 4 + j) * 16) * DM + kk, DM);
      acc[j] = wmma16(a, b, acc[j]);
    }
  }

  // ---------------- LIF1 on accumulators (m = base+r, n = tile + lane%16) --
#pragma unroll
  for (int j = 0; j < 4; ++j) {
    int n = (nh * 4 + j) * 16 + (lane & 15);
#pragma unroll
    for (int r = 0; r < 8; ++r) {
      int mg = mg_base + r;
      float v = first ? 0.0f : v1[(size_t)mg * DS + n];
      v += (acc[j][r] - v) * 0.5f;             // tau = 2, decay_input
      float sp = (v >= 1.0f) ? 1.0f : 0.0f;    // spike(v - vth)
      v = (sp != 0.0f) ? 0.0f : v;             // hard reset
      v1[(size_t)mg * DS + n] = v;
      _Float16 h16 = (_Float16)sp;
      sh_h[(mloc_base + r) * SH_LD + n] = h16; // feeds GEMM2 A-operand
      hbuf[(size_t)mg * DS + n] = h16;         // feeds next timestep
    }
  }
  __syncthreads();

  // ---------------- GEMM2: OU = h_new @ C^T + x_t * Dv  (MT x 512) ---------
#pragma unroll
  for (int blk = 0; blk < 2; ++blk) {
    v8f acc2[8];
#pragma unroll
    for (int j = 0; j < 8; ++j) acc2[j] = vzero;
#pragma unroll
    for (int kk = 0; kk < DS; kk += 32) {
      v16h a = frag_a16(&sh_h[(mi * 16) * SH_LD + kk], SH_LD);
#pragma unroll
      for (int j = 0; j < 8; ++j) {
        int nt = nh * 16 + blk * 8 + j;
        v16h b = frag_b16(Cw + (size_t)(nt * 16) * DS + kk, DS);
        acc2[j] = wmma16(a, b, acc2[j]);
      }
    }
#pragma unroll
    for (int j = 0; j < 8; ++j) {
      int n = (nh * 16 + blk * 8 + j) * 16 + (lane & 15);
      float dv = Dv[n];
#pragma unroll
      for (int r = 0; r < 8; ++r) {
        int mg = mg_base + r;
        float o = acc2[j][r] + (float)xin[(size_t)mg * DM + n] * dv;
        float v = first ? 0.0f : v2[(size_t)mg * DM + n];
        v += (o - v) * 0.5f;
        float sp = (v >= 1.0f) ? 1.0f : 0.0f;
        v = (sp != 0.0f) ? 0.0f : v;
        v2[(size_t)mg * DM + n] = v;
        yout[(size_t)mg * DM + n] = (_Float16)sp;
      }
    }
  }
}

// ---- logits GEMM: counts(2048x512) @ Wp^T(512x32000), *0.05 + bp ----------
// grid (MROWS/64, VOC/256); 8 waves.  Wp tile (256 rows x 32 K, 16KB) staged
// into LDS by the Tensor Data Mover, double buffered; TDM padding inserts
// 16B per 64B row so the LDS stride is 40 f16 (bank-conflict-free b128 reads).
#define LK   32                  /* K slice   */
#define BROW 256                 /* Wp rows per WG tile */
#define BLD  40                  /* padded LDS stride (f16) */
#define NKIT (DM / LK)           /* 16 k iterations */
// group1 word0: data_size=2B (bits17:16=1), pad_enable(20)=1,
// pad_interval(24:22)=3 -> every 16 DWORDs, pad_amount(31:25)=3 -> 4 DWORDs.
#define TDM_FLAGS_PAD 0x06D10000u

__global__ __launch_bounds__(256)
void logits_kernel(const _Float16* __restrict__ cnt,
                   const _Float16* __restrict__ Wp16,   // (VOC, DM) = B-op
                   const float* __restrict__ bp,
                   float* __restrict__ out) {
  __shared__ __align__(16) _Float16 sh_b[2][BROW * BLD];

  const int lane = threadIdx.x & 31;
  const int wv   = threadIdx.x >> 5;
  const int mi   = wv & 3;
  const int nh   = wv >> 2;
  const int m0   = blockIdx.x * 64;
  const int nbase = blockIdx.y * BROW;
  const int mg_base = m0 + mi * 16 + ((lane >> 4) << 3);
  const v8f vzero = {0.f, 0.f, 0.f, 0.f, 0.f, 0.f, 0.f, 0.f};

  v8f acc[8];
#pragma unroll
  for (int j = 0; j < 8; ++j) acc[j] = vzero;

  const _Float16* arow = cnt + (size_t)(m0 + mi * 16) * DM;

#if USE_TDM
  // prologue: wave 0 kicks TDM for k-slice 0
  if (wv == 0)
    tdm_load_2d((unsigned)(uintptr_t)&sh_b[0][0],
                Wp16 + (size_t)nbase * DM, LK, BROW, DM, BROW, DM,
                TDM_FLAGS_PAD);
#else
  // fallback: cooperative staging of slice 0 (16B per thread per step)
  for (int c = threadIdx.x; c < BROW * 4; c += 256) {
    int r = c >> 2, q = c & 3;
    *reinterpret_cast<uint4v*>(&sh_b[0][r * BLD + q * 8]) =
        *reinterpret_cast<const uint4v*>(&Wp16[(size_t)(nbase + r) * DM + q * 8]);
  }
#endif

  for (int kk = 0; kk < NKIT; ++kk) {
    const int cur = kk & 1;
#if USE_TDM
    if (wv == 0) {
      if (kk + 1 < NKIT) {      // prefetch next slice into other buffer
        tdm_load_2d((unsigned)(uintptr_t)&sh_b[cur ^ 1][0],
                    Wp16 + (size_t)nbase * DM + (kk + 1) * LK,
                    LK, BROW, DM, BROW, DM, TDM_FLAGS_PAD);
        __builtin_amdgcn_s_wait_tensorcnt(1);   // current tile complete
      } else {
        __builtin_amdgcn_s_wait_tensorcnt(0);   // last tile complete
      }
    }
#else
    if (kk + 1 < NKIT) {
      for (int c = threadIdx.x; c < BROW * 4; c += 256) {
        int r = c >> 2, q = c & 3;
        *reinterpret_cast<uint4v*>(&sh_b[cur ^ 1][r * BLD + q * 8]) =
            *reinterpret_cast<const uint4v*>(
                &Wp16[(size_t)(nbase + r) * DM + (kk + 1) * LK + q * 8]);
      }
    }
#endif
    __syncthreads();            // buffer `cur` ready for the whole WG

    __builtin_prefetch(arow + (kk + 1) * LK, 0, 0);      // global_prefetch_b8
    v16h a = frag_a16(arow + kk * LK, DM);
#pragma unroll
    for (int j = 0; j < 8; ++j) {
      v16h b = frag_b16(&sh_b[cur][(nh * 8 + j) * 16 * BLD], BLD);
      acc[j] = wmma16(a, b, acc[j]);
    }
    __syncthreads();            // done reading `cur` before it is overwritten
  }

#pragma unroll
  for (int j = 0; j < 8; ++j) {
    int n = nbase + (nh * 8 + j) * 16 + (lane & 15);
    float bb = bp[n];
#pragma unroll
    for (int r = 0; r < 8; ++r) {
      int mg = mg_base + r;
      out[(size_t)mg * VOC + n] = acc[j][r] * 0.05f + bb;   // mean over T=20
    }
  }
}

// ---------------------------------------------------------------------------
extern "C" void kernel_launch(void* const* d_in, const int* in_sizes, int n_in,
                              void* d_out, int out_size, void* d_ws, size_t ws_size,
                              hipStream_t stream) {
  const int*   ids = (const int*)d_in[0];
  const float* emb = (const float*)d_in[1];
  const float* Af  = (const float*)d_in[2];
  const float* Bf  = (const float*)d_in[3];
  const float* Cf  = (const float*)d_in[4];
  const float* Df  = (const float*)d_in[5];
  const float* Wpf = (const float*)d_in[6];
  const float* bpf = (const float*)d_in[7];
  float* out = (float*)d_out;

  char* ws = (char*)d_ws;
  size_t off = 0;
  auto take = [&](size_t bytes) {
    size_t r = off;
    off += (bytes + 255) & ~(size_t)255;
    return r;
  };
  _Float16* A16  = (_Float16*)(ws + take((size_t)NLAYER * DS * DS * 2));
  _Float16* B16  = (_Float16*)(ws + take((size_t)NLAYER * DS * DM * 2));
  _Float16* C16  = (_Float16*)(ws + take((size_t)NLAYER * DM * DS * 2));
  _Float16* Wp16 = (_Float16*)(ws + take((size_t)VOC * DM * 2));
  _Float16* Xb   = (_Float16*)(ws + take((size_t)TS * MROWS * DM * 2));
  _Float16* Yb   = (_Float16*)(ws + take((size_t)TS * MROWS * DM * 2));
  float*    v1   = (float*)   (ws + take((size_t)MROWS * DS * 4));
  float*    v2   = (float*)   (ws + take((size_t)MROWS * DM * 4));
  _Float16* hb   = (_Float16*)(ws + take((size_t)MROWS * DS * 2));
  _Float16* cnt  = (_Float16*)(ws + take((size_t)MROWS * DM * 2));
  (void)ws_size; (void)in_sizes; (void)n_in; (void)out_size;

  // weight conversions (deterministic, re-done every call)
  {
    int n;
    n = NLAYER * DS * DS;
    cvt_f32_f16_kernel<<<(n + 255) / 256, 256, 0, stream>>>(Af, A16, n);
    n = NLAYER * DS * DM;
    cvt_f32_f16_kernel<<<(n + 255) / 256, 256, 0, stream>>>(Bf, B16, n);
    n = NLAYER * DM * DS;
    cvt_f32_f16_kernel<<<(n + 255) / 256, 256, 0, stream>>>(Cf, C16, n);
    n = VOC * DM;
    cvt_f32_f16_kernel<<<(n + 255) / 256, 256, 0, stream>>>(Wpf, Wp16, n);
  }

  // embedding + temporal encode
  encode_kernel<<<(MROWS * DM + 255) / 256, 256, 0, stream>>>(ids, emb, Xb);

  // 4 layers x 20 sequential LIF-SSM steps (state in v1/v2/hb; first=reset)
  for (int l = 0; l < NLAYER; ++l) {
    const _Float16* xin = (l & 1) ? Yb : Xb;
    _Float16*       yut = (l & 1) ? Xb : Yb;
    for (int t = 0; t < TS; ++t) {
      snn_step_kernel<<<MROWS / MT, 256, 0, stream>>>(
          xin + (size_t)t * MROWS * DM, yut + (size_t)t * MROWS * DM,
          A16 + (size_t)l * DS * DS, B16 + (size_t)l * DS * DM,
          C16 + (size_t)l * DM * DS, Df + (size_t)l * DM,
          v1, v2, hb, (t == 0) ? 1 : 0);
    }
  }

  // mean over T (as integer counts) + vocab projection
  reduce_kernel<<<(MROWS * DM + 255) / 256, 256, 0, stream>>>(Xb, cnt);
  dim3 lgrid(MROWS / 64, VOC / 256);
  logits_kernel<<<lgrid, 256, 0, stream>>>(cnt, Wp16, bpf, out);
}
